// SparseFlexAttention_19756849562255
// MI455X (gfx1250) — matrix operand order repaired and checked by reading, not compiled, not run on hardware
//
#include <hip/hip_runtime.h>
#include <math.h>

// ---------------------------------------------------------------------------
// Problem constants (match the JAX reference)
// ---------------------------------------------------------------------------
#define BB 4
#define SS 2048
#define DD 1024

typedef __attribute__((ext_vector_type(16))) __bf16 v16bf;
typedef __attribute__((ext_vector_type(8)))  __bf16 v8bf;
typedef __attribute__((ext_vector_type(8)))  float  v8f;

union FragB16 { v16bf v; v8bf h[2]; };

// A-matrix fragment (16x32, row-major source, row = lane%16 handled by caller):
// lane's 16 bf16 = rows m, k in [8*half, 8*half+8) and [16+8*half, 16+8*half+8)
__device__ __forceinline__ v16bf load_fragA(const __bf16* rowptr, int kblk, int half) {
  FragB16 f;
  f.h[0] = *reinterpret_cast<const v8bf*>(rowptr + kblk + 8 * half);
  f.h[1] = *reinterpret_cast<const v8bf*>(rowptr + kblk + 16 + 8 * half);
  return f.v;
}

// B-matrix fragment (32x16, n-major source, col = lane%16 handled by caller):
// lane's 16 bf16 = k in [8*half, 8*half+8) and [16+8*half, 16+8*half+8)
__device__ __forceinline__ v16bf load_fragB(const __bf16* colptr, int kblk, int half) {
  FragB16 f;
  f.h[0] = *reinterpret_cast<const v8bf*>(colptr + kblk + 8 * half);
  f.h[1] = *reinterpret_cast<const v8bf*>(colptr + kblk + 16 + 8 * half);
  return f.v;
}

__device__ __forceinline__ v8f wmma_bf16(v16bf a, v16bf b, v8f c) {
  // (neg_a, A, neg_b, B, c_mod, C, reuse_a, reuse_b)
  return __builtin_amdgcn_wmma_f32_16x16x32_bf16(false, a, false, b, (short)0, c,
                                                 false, false);
}

// ---------------------------------------------------------------------------
// Kernel 0a: fp32 -> bf16 convert of hidden_states
// ---------------------------------------------------------------------------
__global__ __launch_bounds__(256) void convert_x_kernel(const float* __restrict__ X,
                                                        __bf16* __restrict__ Xbf) {
  size_t i = (size_t)blockIdx.x * 256 + threadIdx.x;
  Xbf[i] = (__bf16)X[i];
}

// ---------------------------------------------------------------------------
// Kernel 0b: transpose+convert the three weight matrices: Wt[n*D + k] = W[k*D + n]
// (n-major storage so B-fragments are contiguous b128 loads)
// ---------------------------------------------------------------------------
__global__ __launch_bounds__(256) void transpose_w_kernel(const float* __restrict__ Wq,
                                                          const float* __restrict__ Wk,
                                                          const float* __restrict__ Wv,
                                                          __bf16* __restrict__ Wt) {
  const int z = blockIdx.z;
  const float* W = (z == 0) ? Wq : (z == 1) ? Wk : Wv;
  __bf16* out = Wt + (size_t)z * DD * DD;
  size_t i = (size_t)blockIdx.x * 256 + threadIdx.x;  // i = k*D + n
  int k = (int)(i >> 10);
  int n = (int)(i & (DD - 1));
  out[(size_t)n * DD + k] = (__bf16)W[i];
}

// ---------------------------------------------------------------------------
// Kernel 1: fused QKV projection. C[8192,1024] = Xbf @ W (+bias).
// Block = 256 threads (8 waves) arranged 2(M) x 4(N); block tile 128 x 256.
// Wave tile 64x64 = 4x4 WMMA accumulators -> 16 WMMA per 16 b128 loads per
// K-step (issue-slot balanced; everything is L2-resident on the 192MB L2).
// z selects Q / K / V; V is written TRANSPOSED (Vt[b][d][s]) for the PV stage.
// ---------------------------------------------------------------------------
__global__ __launch_bounds__(256) void qkv_gemm_kernel(
    const __bf16* __restrict__ Xbf, const __bf16* __restrict__ Wt,
    const float* __restrict__ bq, const float* __restrict__ bk,
    const float* __restrict__ bv, __bf16* __restrict__ Qb,
    __bf16* __restrict__ Kb, __bf16* __restrict__ Vt) {
  const int tid = threadIdx.x;
  const int lane = tid & 31, wave = tid >> 5;
  const int half = lane >> 4, l16 = lane & 15;
  const int z = blockIdx.z;
  const int m0 = blockIdx.y * 128 + (wave & 1) * 64;   // 2 waves along M
  const int n0 = blockIdx.x * 256 + (wave >> 1) * 64;  // 4 waves along N

  const __bf16* Ap[4];
  const __bf16* Bp[4];
  const __bf16* Wz = Wt + (size_t)z * DD * DD;
#pragma unroll
  for (int i = 0; i < 4; ++i) {
    Ap[i] = Xbf + (size_t)(m0 + i * 16 + l16) * DD;
    Bp[i] = Wz + (size_t)(n0 + i * 16 + l16) * DD;
  }

  v8f acc[4][4] = {};
  for (int k = 0; k < DD; k += 32) {
    v16bf a[4], b[4];
#pragma unroll
    for (int i = 0; i < 4; ++i) a[i] = load_fragA(Ap[i], k, half);
#pragma unroll
    for (int i = 0; i < 4; ++i) b[i] = load_fragB(Bp[i], k, half);
    if (k + 32 < DD) {  // pull next K-block toward the WGP while WMMAs drain
      __builtin_prefetch(Ap[0] + k + 32, 0, 1);
      __builtin_prefetch(Bp[0] + k + 32, 0, 1);
    }
#pragma unroll
    for (int mi = 0; mi < 4; ++mi)
#pragma unroll
      for (int ni = 0; ni < 4; ++ni) acc[mi][ni] = wmma_bf16(a[mi], b[ni], acc[mi][ni]);
  }

  const float* bias = (z == 0) ? bq : (z == 1) ? bk : bv;
  __bf16* out = (z == 0) ? Qb : Kb;
#pragma unroll
  for (int mi = 0; mi < 4; ++mi)
#pragma unroll
    for (int ni = 0; ni < 4; ++ni)
#pragma unroll
      for (int r = 0; r < 8; ++r) {
        int m = m0 + mi * 16 + r + 8 * half;  // C layout: M = r (+8 upper half)
        int n = n0 + ni * 16 + l16;           // N = lane%16
        float v = acc[mi][ni][r] + bias[n];
        if (z < 2) {
          out[(size_t)m * DD + n] = (__bf16)v;
        } else {
          int bidx = m >> 11;        // m / S
          int s = m & (SS - 1);      // m % S
          Vt[((size_t)bidx * DD + n) * SS + s] = (__bf16)v;  // transposed V
        }
      }
}

// ---------------------------------------------------------------------------
// Kernel 2: fused attention for one (batch, 32-query-row) tile.
//   Phase 1: scores = Q K^T / 32, sparse-mask -> -inf, + attention_mask,
//            stored bf16 into LDS [32][2056]. 2 groups of 2x8 accumulators
//            per wave so Q fragments are reused across 8 key subtiles.
//   Phase 2: softmax (8 threads per row), exp written back in place (bf16),
//            per-row 1/sum kept in LDS.
//   Phase 3: context = P @ V via WMMA (A-frags from LDS, B-frags from global Vt),
//            row-rescaled by 1/sum in the epilogue.
// ---------------------------------------------------------------------------
#define LP 2056  // padded row stride (bf16 elems): 4112 B, 16B aligned, de-banked

__global__ __launch_bounds__(256) void attn_kernel(
    const __bf16* __restrict__ Qb, const __bf16* __restrict__ Kb,
    const __bf16* __restrict__ Vt, const unsigned char* __restrict__ smask,
    const float* __restrict__ amask, float* __restrict__ out) {
  extern __shared__ char smem_raw[];
  __bf16* probs = (__bf16*)smem_raw;                                   // [32][LP]
  float* red = (float*)(smem_raw + (size_t)32 * LP * sizeof(__bf16));  // [32][8]
  float* invsum = red + 32 * 8;                                        // [32]

  const int tid = threadIdx.x, lane = tid & 31, wave = tid >> 5;
  const int half = lane >> 4, l16 = lane & 15;
  const int q0 = blockIdx.x * 32;
  const int b = blockIdx.y;

  // ---- Phase 1: scores tile (each wave owns 256 key columns, 2 groups) ----
  const __bf16* Q0 = Qb + ((size_t)b * SS + q0 + l16) * DD;
  const __bf16* Q1 = Q0 + (size_t)16 * DD;
  const float scale = 0.03125f;  // 1/sqrt(1024)
#pragma unroll
  for (int g = 0; g < 2; ++g) {
    const int nb0 = wave * 256 + g * 128;
    v8f acc[2][8] = {};
    for (int d = 0; d < DD; d += 32) {
      v16bf a0 = load_fragA(Q0, d, half);
      v16bf a1 = load_fragA(Q1, d, half);
      if (d + 32 < DD) __builtin_prefetch(Q0 + d + 32, 0, 1);
#pragma unroll
      for (int ns = 0; ns < 8; ++ns) {
        const __bf16* Kp = Kb + ((size_t)b * SS + nb0 + ns * 16 + l16) * DD;
        v16bf bf = load_fragB(Kp, d, half);
        acc[0][ns] = wmma_bf16(a0, bf, acc[0][ns]);
        acc[1][ns] = wmma_bf16(a1, bf, acc[1][ns]);
      }
    }
#pragma unroll
    for (int ms = 0; ms < 2; ++ms)
#pragma unroll
      for (int ns = 0; ns < 8; ++ns)
#pragma unroll
        for (int r = 0; r < 8; ++r) {
          int m = ms * 16 + r + 8 * half;
          int n = nb0 + ns * 16 + l16;
          float s = acc[ms][ns][r] * scale;
          if (smask[((size_t)b * SS + q0 + m) * SS + n]) s = -__builtin_inff();
          s += amask[(size_t)b * SS + n];
          probs[m * LP + n] = (__bf16)s;
        }
  }
  __syncthreads();

  // ---- Phase 2: softmax (8 threads per row, 256 cols each) ----
  {
    const int row = tid >> 3, seg = tid & 7;
    __bf16* prow = probs + row * LP + seg * 256;
    float mx = -__builtin_inff();
    for (int j = 0; j < 256; ++j) mx = fmaxf(mx, (float)prow[j]);
    red[row * 8 + seg] = mx;
    __syncthreads();
    float rmax = red[row * 8];
#pragma unroll
    for (int j = 1; j < 8; ++j) rmax = fmaxf(rmax, red[row * 8 + j]);
    __syncthreads();  // everyone done reading maxima before red is reused
    float sum = 0.f;
    for (int j = 0; j < 256; ++j) {
      float p = __expf((float)prow[j] - rmax);
      sum += p;
      prow[j] = (__bf16)p;  // un-normalized prob, bf16, in place
    }
    red[row * 8 + seg] = sum;
    __syncthreads();
    if (seg == 0) {
      float s = 0.f;
#pragma unroll
      for (int j = 0; j < 8; ++j) s += red[row * 8 + j];
      invsum[row] = 1.0f / s;
    }
  }
  __syncthreads();

  // ---- Phase 3: context = P @ V (each wave owns 128 d-columns) ----
  const int nc0 = wave * 128;
  v8f acc2[2][8] = {};
  for (int kk = 0; kk < SS; kk += 32) {
    v16bf a0, a1;
    {
      FragB16 f;
      f.h[0] = *reinterpret_cast<const v8bf*>(probs + l16 * LP + kk + 8 * half);
      f.h[1] = *reinterpret_cast<const v8bf*>(probs + l16 * LP + kk + 16 + 8 * half);
      a0 = f.v;
      f.h[0] = *reinterpret_cast<const v8bf*>(probs + (16 + l16) * LP + kk + 8 * half);
      f.h[1] = *reinterpret_cast<const v8bf*>(probs + (16 + l16) * LP + kk + 16 + 8 * half);
      a1 = f.v;
    }
#pragma unroll
    for (int ns = 0; ns < 8; ++ns) {
      const __bf16* Vcol = Vt + ((size_t)b * DD + nc0 + ns * 16 + l16) * SS;
      v16bf bf = load_fragB(Vcol, kk, half);
      acc2[0][ns] = wmma_bf16(a0, bf, acc2[0][ns]);
      acc2[1][ns] = wmma_bf16(a1, bf, acc2[1][ns]);
    }
  }
#pragma unroll
  for (int ms = 0; ms < 2; ++ms)
#pragma unroll
    for (int ns = 0; ns < 8; ++ns)
#pragma unroll
      for (int r = 0; r < 8; ++r) {
        int m = ms * 16 + r + 8 * half;
        int n = nc0 + ns * 16 + l16;
        out[((size_t)b * SS + q0 + m) * DD + n] = acc2[ms][ns][r] * invsum[m];
      }
}

// ---------------------------------------------------------------------------
// Host launcher
// ---------------------------------------------------------------------------
extern "C" void kernel_launch(void* const* d_in, const int* in_sizes, int n_in,
                              void* d_out, int out_size, void* d_ws, size_t ws_size,
                              hipStream_t stream) {
  const float* X = (const float*)d_in[0];
  const float* Wq = (const float*)d_in[1];
  const float* bq = (const float*)d_in[2];
  const float* Wk = (const float*)d_in[3];
  const float* bk = (const float*)d_in[4];
  const float* Wv = (const float*)d_in[5];
  const float* bv = (const float*)d_in[6];
  const unsigned char* smask = (const unsigned char*)d_in[7];  // jnp bool = 1 byte
  const float* amask = (const float*)d_in[8];
  float* out = (float*)d_out;

  // workspace layout (bf16): Xbf | Wt(x3) | Q | K | Vt  ~= 70 MB
  __bf16* Xbf = (__bf16*)d_ws;
  __bf16* Wt = Xbf + (size_t)BB * SS * DD;
  __bf16* Qb = Wt + (size_t)3 * DD * DD;
  __bf16* Kb = Qb + (size_t)BB * SS * DD;
  __bf16* Vt = Kb + (size_t)BB * SS * DD;

  const size_t smem_bytes = (size_t)32 * LP * sizeof(__bf16) + 32 * 8 * sizeof(float) +
                            32 * sizeof(float);  // ~132 KB, CDNA5 WGP has 320 KB
  // opt in to >64KB dynamic LDS (attribute set, not a stream op; capture-safe)
  hipFuncSetAttribute((const void*)attn_kernel,
                      hipFuncAttributeMaxDynamicSharedMemorySize, (int)smem_bytes);

  convert_x_kernel<<<dim3((BB * SS * DD) / 256), 256, 0, stream>>>(X, Xbf);
  transpose_w_kernel<<<dim3((DD * DD) / 256, 1, 3), 256, 0, stream>>>(Wq, Wk, Wv, Wt);
  qkv_gemm_kernel<<<dim3(DD / 256, (BB * SS) / 128, 3), 256, 0, stream>>>(
      Xbf, Wt, bq, bk, bv, Qb, Kb, Vt);
  attn_kernel<<<dim3(SS / 32, BB), 256, smem_bytes, stream>>>(Qb, Kb, Vt, smask,
                                                              amask, out);
  (void)in_sizes; (void)n_in; (void)out_size; (void)ws_size;
}
